// Mult_Head_Inten_53893249630275
// MI455X (gfx1250) — matrix operand order repaired and checked by reading, not compiled, not run on hardware
//
#include <hip/hip_runtime.h>
#include <hip/hip_bf16.h>

// ---------- types ----------
typedef __bf16 v16bf __attribute__((ext_vector_type(16)));
typedef float  v8f   __attribute__((ext_vector_type(8)));

union FragU { uint4 u[2]; v16bf v; };

// float -> bf16 (round to nearest even)
__device__ __forceinline__ unsigned short f2bf(float f) {
  union { float f; unsigned u; } x; x.f = f;
  unsigned r = x.u + 0x7FFFu + ((x.u >> 16) & 1u);
  return (unsigned short)(r >> 16);
}

// A fragment: 16x32 bf16, row-major src, leading dim ld (elements).
// Lane layout (ISA 7.12.2): row = lane&15; K base = (lane>>4)*8;
// vgpr0..3 hold K=kb..kb+7, vgpr4..7 hold K=kb+16..kb+23.
__device__ __forceinline__ v16bf load_frag_a(const unsigned short* p, int ld,
                                             int row0, int k0, int lane) {
  int m = lane & 15, kb = (lane >> 4) * 8;
  const unsigned short* q = p + (size_t)(row0 + m) * ld + k0 + kb;
  FragU f;
  f.u[0] = *(const uint4*)(q);
  f.u[1] = *(const uint4*)(q + 16);
  return f.v;
}

// B fragment: 32x16 bf16 from N-major storage bt[n][k], leading dim ld.
// Lane layout: col n = lane&15; K base = (lane>>4)*8; K=kb..kb+7, kb+16..kb+23.
__device__ __forceinline__ v16bf load_frag_b(const unsigned short* bt, int ld,
                                             int n0, int k0, int lane) {
  int n = lane & 15, kb = (lane >> 4) * 8;
  const unsigned short* q = bt + (size_t)(n0 + n) * ld + k0 + kb;
  FragU f;
  f.u[0] = *(const uint4*)(q);
  f.u[1] = *(const uint4*)(q + 16);
  return f.v;
}

__device__ __forceinline__ v8f wmma_bf16(v16bf a, v16bf b, v8f c) {
  return __builtin_amdgcn_wmma_f32_16x16x32_bf16(false, a, false, b,
                                                 (short)0, c, false, false);
}

// 16-lane butterfly reductions (rows of a C-fragment live in one 16-lane half)
__device__ __forceinline__ float hr_max(float v) {
#pragma unroll
  for (int off = 1; off < 16; off <<= 1) v = fmaxf(v, __shfl_xor(v, off, 32));
  return v;
}
__device__ __forceinline__ float hr_sum(float v) {
#pragma unroll
  for (int off = 1; off < 16; off <<= 1) v += __shfl_xor(v, off, 32);
  return v;
}

// ---------- problem constants ----------
#define BB 4
#define NN 2048
#define CC 1024
#define HH 16
#define DD 64
#define ROWS (BB * NN)   // 8192

// ---------- shared GEMM mainloop: 32x64 per wave, K=1024, double-buffered ----
// acc[sub][t]: rows m0+16*sub, cols n0+16*t.
__device__ __forceinline__ void gemm_mainloop_32x64(
    const unsigned short* __restrict__ A, const unsigned short* __restrict__ Bt,
    int m0, int n0, int lane, v8f acc[2][4]) {
  v16bf a0 = load_frag_a(A, CC, m0, 0, lane);
  v16bf a1 = load_frag_a(A, CC, m0 + 16, 0, lane);
  v16bf b0 = load_frag_b(Bt, CC, n0, 0, lane);
  v16bf b1 = load_frag_b(Bt, CC, n0 + 16, 0, lane);
  v16bf b2 = load_frag_b(Bt, CC, n0 + 32, 0, lane);
  v16bf b3 = load_frag_b(Bt, CC, n0 + 48, 0, lane);
  for (int kk = 0; kk < CC - 32; kk += 32) {
    int kn = kk + 32;
    // prefetch next k-step fragments (overlaps with this step's WMMAs)
    v16bf na0 = load_frag_a(A, CC, m0, kn, lane);
    v16bf na1 = load_frag_a(A, CC, m0 + 16, kn, lane);
    v16bf nb0 = load_frag_b(Bt, CC, n0, kn, lane);
    v16bf nb1 = load_frag_b(Bt, CC, n0 + 16, kn, lane);
    v16bf nb2 = load_frag_b(Bt, CC, n0 + 32, kn, lane);
    v16bf nb3 = load_frag_b(Bt, CC, n0 + 48, kn, lane);
    acc[0][0] = wmma_bf16(a0, b0, acc[0][0]);
    acc[1][0] = wmma_bf16(a1, b0, acc[1][0]);
    acc[0][1] = wmma_bf16(a0, b1, acc[0][1]);
    acc[1][1] = wmma_bf16(a1, b1, acc[1][1]);
    acc[0][2] = wmma_bf16(a0, b2, acc[0][2]);
    acc[1][2] = wmma_bf16(a1, b2, acc[1][2]);
    acc[0][3] = wmma_bf16(a0, b3, acc[0][3]);
    acc[1][3] = wmma_bf16(a1, b3, acc[1][3]);
    a0 = na0; a1 = na1; b0 = nb0; b1 = nb1; b2 = nb2; b3 = nb3;
  }
  acc[0][0] = wmma_bf16(a0, b0, acc[0][0]);
  acc[1][0] = wmma_bf16(a1, b0, acc[1][0]);
  acc[0][1] = wmma_bf16(a0, b1, acc[0][1]);
  acc[1][1] = wmma_bf16(a1, b1, acc[1][1]);
  acc[0][2] = wmma_bf16(a0, b2, acc[0][2]);
  acc[1][2] = wmma_bf16(a1, b2, acc[1][2]);
  acc[0][3] = wmma_bf16(a0, b3, acc[0][3]);
  acc[1][3] = wmma_bf16(a1, b3, acc[1][3]);
}

// ---------- kernel 0: x fp32 -> bf16 ----------
__global__ void k_cvt_x(const float* __restrict__ x,
                        unsigned short* __restrict__ xb, int n) {
  int i = (blockIdx.x * blockDim.x + threadIdx.x) * 4;
  if (i >= n) return;
  float4 v = *(const float4*)(x + i);
  uint2 pk;
  pk.x = (unsigned)f2bf(v.x) | ((unsigned)f2bf(v.y) << 16);
  pk.y = (unsigned)f2bf(v.z) | ((unsigned)f2bf(v.w) << 16);
  *(uint2*)(xb + i) = pk;
}

// ---------- kernel 1: W [K][Nn] fp32 -> Wt [Nn][K] bf16 ----------
__global__ void k_transpose_bf16(const float* __restrict__ w,
                                 unsigned short* __restrict__ wt,
                                 int K, int Nn) {
  int idx = blockIdx.x * blockDim.x + threadIdx.x;
  if (idx >= K * Nn) return;
  int k = idx / Nn, n = idx - k * Nn;
  wt[(size_t)n * K + k] = f2bf(w[idx]);
}

// ---------- kernel 2: QKV GEMM (8192 x 3072 x 1024) ----------
// wave: 32x64 tile, 8 waves/block -> 256 rows x 64 cols per block.
// Epilogue: +bias, scale Q by 1/8, scatter to Q[B,H,N,D], K[B,H,N,D], Vt[B,H,D,N].
__global__ __launch_bounds__(256) void k_qkv_gemm(
    const unsigned short* __restrict__ xb, const unsigned short* __restrict__ wT,
    const float* __restrict__ bias, unsigned short* __restrict__ qb,
    unsigned short* __restrict__ kb, unsigned short* __restrict__ vT) {
  int lane = threadIdx.x & 31, wave = threadIdx.x >> 5;
  int m0 = blockIdx.x * 256 + wave * 32;
  int n0 = blockIdx.y * 64;
  v8f acc[2][4] = {};
  gemm_mainloop_32x64(xb, wT, m0, n0, lane, acc);

  int hw = lane >> 4, cn = lane & 15;
#pragma unroll
  for (int t = 0; t < 4; ++t) {
    int col = n0 + 16 * t + cn;
    int which = col >> 10;           // 0=q 1=k 2=v
    int rem = col & (CC - 1);
    int head = rem >> 6, d = rem & (DD - 1);
    float bv = bias[col];
#pragma unroll
    for (int sub = 0; sub < 2; ++sub) {
#pragma unroll
      for (int r = 0; r < 8; ++r) {
        int row = m0 + 16 * sub + r + 8 * hw;
        int bidx = row >> 11, nn = row & (NN - 1);
        int bh = bidx * HH + head;
        float v = acc[sub][t][r] + bv;
        if (which == 0)
          qb[((size_t)bh * NN + nn) * DD + d] = f2bf(v * 0.125f);  // 1/sqrt(64)
        else if (which == 1)
          kb[((size_t)bh * NN + nn) * DD + d] = f2bf(v);
        else
          vT[((size_t)bh * DD + d) * NN + nn] = f2bf(v);
      }
    }
  }
}

// ---------- kernel 3: flash attention per 16-query tile ----------
// wave owns one (b,h,qtile). 32-key chunks: 4 WMMAs for scores (Kdim=64),
// online softmax, P staged through LDS (C-layout -> A-layout), 4 WMMAs for P@V.
__global__ __launch_bounds__(128) void k_attn(
    const unsigned short* __restrict__ qb, const unsigned short* __restrict__ kb,
    const unsigned short* __restrict__ vT, unsigned short* __restrict__ attnb) {
  __shared__ unsigned short psmem[4][16 * 32];
  int lane = threadIdx.x & 31, wave = threadIdx.x >> 5;
  int gw = blockIdx.x * 4 + wave;
  int qt = gw & (NN / 16 - 1);   // 128 q-tiles
  int bh = gw >> 7;              // 0..63
  const unsigned short* Q = qb + (size_t)bh * NN * DD;
  const unsigned short* K = kb + (size_t)bh * NN * DD;
  const unsigned short* V = vT + (size_t)bh * DD * NN;
  int row0 = qt * 16;

  v16bf aq0 = load_frag_a(Q, DD, row0, 0, lane);
  v16bf aq1 = load_frag_a(Q, DD, row0, 32, lane);

  float mrow[8], lrow[8];
#pragma unroll
  for (int r = 0; r < 8; ++r) { mrow[r] = -3.0e38f; lrow[r] = 0.f; }
  v8f o[4] = {};
  unsigned short* ps = &psmem[wave][0];
  int hw = lane >> 4, cn = lane & 15;

  for (int kv = 0; kv < NN; kv += 32) {
    // all fragment loads for this chunk issue up front (overlap with softmax)
    v16bf kb0 = load_frag_b(K, DD, kv, 0, lane);
    v16bf kb1 = load_frag_b(K, DD, kv, 32, lane);
    v16bf kb2 = load_frag_b(K, DD, kv + 16, 0, lane);
    v16bf kb3 = load_frag_b(K, DD, kv + 16, 32, lane);
    v16bf vb0 = load_frag_b(V, NN, 0, kv, lane);
    v16bf vb1 = load_frag_b(V, NN, 16, kv, lane);
    v16bf vb2 = load_frag_b(V, NN, 32, kv, lane);
    v16bf vb3 = load_frag_b(V, NN, 48, kv, lane);
    // pull next chunk of K toward the caches (global_prefetch_b8)
    if (kv + 32 < NN)
      __builtin_prefetch(K + ((size_t)(kv + 32)) * DD + lane * 64, 0, 1);

    // scores: S[16q x 32k] (scale already folded into Q)
    v8f s0 = {}, s1 = {};
    s0 = wmma_bf16(aq0, kb0, s0);
    s0 = wmma_bf16(aq1, kb1, s0);
    s1 = wmma_bf16(aq0, kb2, s1);
    s1 = wmma_bf16(aq1, kb3, s1);

    float corr[8];
#pragma unroll
    for (int r = 0; r < 8; ++r) {
      float rm = hr_max(fmaxf(s0[r], s1[r]));
      float mn = fmaxf(mrow[r], rm);
      float p0 = __expf(s0[r] - mn);
      float p1 = __expf(s1[r] - mn);
      s0[r] = p0; s1[r] = p1;
      float rs = hr_sum(p0 + p1);
      corr[r] = __expf(mrow[r] - mn);
      lrow[r] = lrow[r] * corr[r] + rs;
      mrow[r] = mn;
    }
#pragma unroll
    for (int t = 0; t < 4; ++t)
#pragma unroll
      for (int r = 0; r < 8; ++r) o[t][r] *= corr[r];

    // P: C-layout -> LDS -> A-layout
#pragma unroll
    for (int r = 0; r < 8; ++r) {
      int row = r + 8 * hw;
      ps[row * 32 + cn]      = f2bf(s0[r]);
      ps[row * 32 + 16 + cn] = f2bf(s1[r]);
    }
    asm volatile("s_wait_dscnt 0" ::: "memory");
    v16bf pa = load_frag_a(ps, 32, 0, 0, lane);

    // O += P @ V  (V^T is N-major: B-frag loads contiguous along keys)
    o[0] = wmma_bf16(pa, vb0, o[0]);
    o[1] = wmma_bf16(pa, vb1, o[1]);
    o[2] = wmma_bf16(pa, vb2, o[2]);
    o[3] = wmma_bf16(pa, vb3, o[3]);
    asm volatile("s_wait_dscnt 0" ::: "memory");
  }

  // finalize + store attn in [B, N, C] bf16 for proj GEMM
  int b = bh >> 4, h = bh & (HH - 1);
#pragma unroll
  for (int r = 0; r < 8; ++r) {
    float inv = 1.0f / lrow[r];
    int row = row0 + r + 8 * hw;
#pragma unroll
    for (int t = 0; t < 4; ++t)
      attnb[(size_t)(b * NN + row) * CC + h * DD + 16 * t + cn] =
          f2bf(o[t][r] * inv);
  }
}

// ---------- kernel 4: proj GEMM (8192 x 1024 x 1024), fp32 out + bias ----------
__global__ __launch_bounds__(256) void k_proj_gemm(
    const unsigned short* __restrict__ ab, const unsigned short* __restrict__ wT,
    const float* __restrict__ bias, float* __restrict__ out) {
  int lane = threadIdx.x & 31, wave = threadIdx.x >> 5;
  int m0 = blockIdx.x * 256 + wave * 32;
  int n0 = blockIdx.y * 64;
  v8f acc[2][4] = {};
  gemm_mainloop_32x64(ab, wT, m0, n0, lane, acc);

  int hw = lane >> 4, cn = lane & 15;
#pragma unroll
  for (int t = 0; t < 4; ++t) {
    int col = n0 + 16 * t + cn;
    float bv = bias[col];
#pragma unroll
    for (int sub = 0; sub < 2; ++sub) {
#pragma unroll
      for (int r = 0; r < 8; ++r) {
        int row = m0 + 16 * sub + r + 8 * hw;
        out[(size_t)row * CC + col] = acc[sub][t][r] + bv;
      }
    }
  }
}

// ---------- launch ----------
extern "C" void kernel_launch(void* const* d_in, const int* in_sizes, int n_in,
                              void* d_out, int out_size, void* d_ws, size_t ws_size,
                              hipStream_t stream) {
  (void)in_sizes; (void)n_in; (void)out_size; (void)ws_size;
  const float* x      = (const float*)d_in[0];
  const float* W_qkv  = (const float*)d_in[1];
  const float* b_qkv  = (const float*)d_in[2];
  const float* W_proj = (const float*)d_in[3];
  const float* b_proj = (const float*)d_in[4];
  float* out = (float*)d_out;

  // workspace carve (256B aligned regions)
  unsigned char* w = (unsigned char*)d_ws;
  size_t off = 0;
  auto carve = [&](size_t bytes) {
    unsigned char* p = w + off;
    off += (bytes + 255) & ~(size_t)255;
    return p;
  };
  unsigned short* xb     = (unsigned short*)carve((size_t)ROWS * CC * 2);      // 16 MiB
  unsigned short* wqkvT  = (unsigned short*)carve((size_t)3 * CC * CC * 2);    //  6 MiB
  unsigned short* wprojT = (unsigned short*)carve((size_t)CC * CC * 2);        //  2 MiB
  unsigned short* qb     = (unsigned short*)carve((size_t)ROWS * CC * 2);      // 16 MiB
  unsigned short* kb     = (unsigned short*)carve((size_t)ROWS * CC * 2);      // 16 MiB
  unsigned short* vT     = (unsigned short*)carve((size_t)ROWS * CC * 2);      // 16 MiB
  unsigned short* attnb  = xb;  // xb dead after QKV GEMM -> reuse for attn out

  // 0) x -> bf16
  {
    int n = ROWS * CC;  // 8,388,608
    k_cvt_x<<<n / (256 * 4), 256, 0, stream>>>(x, xb, n);
  }
  // 1) weight transposes -> bf16, N-major
  {
    int n1 = CC * 3 * CC;
    k_transpose_bf16<<<(n1 + 255) / 256, 256, 0, stream>>>(W_qkv, wqkvT, CC, 3 * CC);
    int n2 = CC * CC;
    k_transpose_bf16<<<(n2 + 255) / 256, 256, 0, stream>>>(W_proj, wprojT, CC, CC);
  }
  // 2) QKV GEMM
  k_qkv_gemm<<<dim3(ROWS / 256, (3 * CC) / 64), 256, 0, stream>>>(
      xb, wqkvT, b_qkv, qb, kb, vT);
  // 3) attention (B*H*(N/16) waves / 4 waves per block)
  k_attn<<<(BB * HH * (NN / 16)) / 4, 128, 0, stream>>>(qb, kb, vT, attnb);
  // 4) projection
  k_proj_gemm<<<dim3(ROWS / 256, CC / 64), 256, 0, stream>>>(
      attnb, wprojT, b_proj, out);
}
